// Tree_Batch_Mono_GLM_59236188946459
// MI455X (gfx1250) — matrix operand order repaired and checked by reading, not compiled, not run on hardware
//
#include <hip/hip_runtime.h>
#include <hip/hip_bf16.h>

#define SUB_NO 20
#define E_NO   2000
#define I_NO   500
#define T_NO   200
#define BATCH  8
#define T_DATA 5000

#define KCH_E  63   // ceil(2000/32)
#define KCH_I  16   // ceil(500/32)

typedef __attribute__((ext_vector_type(16))) _Float16 v16h;
typedef __attribute__((ext_vector_type(8)))  float    v8f;

// ---------------------------------------------------------------------------
// Column softmax over the 20 subunits + pack the result into WMMA A-fragment
// layout (16-bit A 16x32 in this code's own contiguous K order, same as B):
//   lane L (row M = L&15, half = L>>4), element e: K = 16*half+e
// afrag[rt][kc][lane][elem], rt = row-tile (s 0..15 / 16..31, rows>=20 zero)
// ---------------------------------------------------------------------------
__global__ void k_softmax_afrag(const float* __restrict__ logits,
                                const int* __restrict__ temp_i,
                                float* __restrict__ probs_out,
                                _Float16* __restrict__ afrag,
                                int K, int KCH)
{
    int e = blockIdx.x * blockDim.x + threadIdx.x;
    int epad = KCH * 32;
    if (e >= epad) return;

    float p[SUB_NO];
    bool valid = (e < K);
    if (valid) {
        float invT = 1.0f / (float)temp_i[0];
        float mx = -1e30f;
#pragma unroll
        for (int s = 0; s < SUB_NO; ++s) {
            p[s] = logits[s * K + e] * invT;
            mx = fmaxf(mx, p[s]);
        }
        float sum = 0.f;
#pragma unroll
        for (int s = 0; s < SUB_NO; ++s) { p[s] = expf(p[s] - mx); sum += p[s]; }
        float r = 1.0f / sum;
#pragma unroll
        for (int s = 0; s < SUB_NO; ++s) { p[s] *= r; probs_out[s * K + e] = p[s]; }
    }

    int kc   = e >> 5;
    int ek   = e & 31;
    int lh   = ek >> 4;                             // which lane-half holds this K
    int elem = ek & 15;
#pragma unroll
    for (int s2 = 0; s2 < 32; ++s2) {
        int rt = s2 >> 4, sl = s2 & 15;
        int lane = sl + 16 * lh;
        float v = (valid && (s2 < SUB_NO)) ? p[s2] : 0.0f;
        afrag[((size_t)(rt * KCH + kc) * 32 + lane) * 16 + elem] = (_Float16)v;
    }
}

// ---------------------------------------------------------------------------
// C_den (tree softmax), coupling matrix M[j][k] = C_den[j][k]*exp(W_sub[k]),
// and the tiled out_filters output. One block.
// ---------------------------------------------------------------------------
__global__ void k_setup_den(const float* __restrict__ raw,
                            const float* __restrict__ W_syn,
                            const float* __restrict__ W_sub,
                            const int* __restrict__ temp_i,
                            float* __restrict__ cden_out,   // [20][20]
                            float* __restrict__ filt_out,   // [40][200]
                            float* __restrict__ Mmat)       // [20][20]
{
    int tid = threadIdx.x;
    float invT = 1.0f / (float)temp_i[0];

    if (tid < SUB_NO) {
        int j = tid;
        float col[SUB_NO];
        for (int r = 0; r < SUB_NO; ++r) col[r] = 0.0f;
        if (j == 1) col[0] = 1.0f;                       // C_den[0,1] = 1
        if (j >= 2) {
            int off = j * (j - 1) / 2 - 1;               // segment offset
            float mx = -1e30f;
            for (int r = 0; r < j; ++r) {
                float z = raw[off + r] * invT;
                col[r] = z;
                mx = fmaxf(mx, z);
            }
            float sum = 0.f;
            for (int r = 0; r < j; ++r) { col[r] = expf(col[r] - mx); sum += col[r]; }
            float rs = 1.0f / sum;
            for (int r = 0; r < j; ++r) col[r] *= rs;    // rows >= j stay 0
        }
        for (int r = 0; r < SUB_NO; ++r) cden_out[r * SUB_NO + j] = col[r];
    }
    __syncthreads();

    for (int i = tid; i < SUB_NO * SUB_NO; i += blockDim.x) {
        int k = i % SUB_NO;
        Mmat[i] = cden_out[i] * expf(W_sub[k]);
    }
    for (int i = tid; i < 2 * SUB_NO * T_NO; i += blockDim.x) {
        int row = i / T_NO, m = i % T_NO;
        filt_out[i] = W_syn[2 * m + (row >= SUB_NO ? 1 : 0)];
    }
}

// ---------------------------------------------------------------------------
// syn[b][t][s] = sum_e C[s,e] * S[b,t,e] via v_wmma_f32_16x16x32_f16.
// One wave per (b, 16-wide t tile). A = weight fragments (pre-swizzled),
// B per lane: column N = t0+(lane&15), K = kc*32 + 16*(lane>>4) + [0..16).
// D layout: VGPR r, lane L -> M = r + 8*(L>>4), N = L&15.
// K/KCH are compile-time so the steady-state loop has no bounds checks;
// only the final partial chunk (K%32 != 0) is masked, once, outside the loop.
// ---------------------------------------------------------------------------
template<int K, int KCH>
__global__ void k_gemm_syn(const float* __restrict__ S,
                           const _Float16* __restrict__ afrag,
                           float* __restrict__ out)
{
    constexpr int FULL = (K % 32 == 0) ? KCH : (KCH - 1); // chunks with all 32 K valid

    int lane = threadIdx.x;
    int lh = lane >> 4, nl = lane & 15;
    int t0 = blockIdx.x * 16;
    int b  = blockIdx.y;
    int t  = t0 + nl;
    int tc = (t < T_DATA) ? t : (T_DATA - 1);
    const float* src = S + (size_t)(b * T_DATA + tc) * K + lh * 16;
    const _Float16* ap0 = afrag + (size_t)lane * 16;
    const _Float16* ap1 = afrag + ((size_t)KCH * 32 + lane) * 16;

    v8f acc0 = {}, acc1 = {};

#pragma unroll 2
    for (int kc = 0; kc < FULL; ++kc) {
        if (kc * 32 + 512 < K)                       // wave-uniform predicate
            __builtin_prefetch(src + kc * 32 + 512, 0, 1);
        const float* sp = src + kc * 32;
        float4 v4a = *(const float4*)(sp);
        float4 v4b = *(const float4*)(sp + 4);
        float4 v4c = *(const float4*)(sp + 8);
        float4 v4d = *(const float4*)(sp + 12);
        v16h bfrag;
        bfrag[0]  = (_Float16)v4a.x; bfrag[1]  = (_Float16)v4a.y;
        bfrag[2]  = (_Float16)v4a.z; bfrag[3]  = (_Float16)v4a.w;
        bfrag[4]  = (_Float16)v4b.x; bfrag[5]  = (_Float16)v4b.y;
        bfrag[6]  = (_Float16)v4b.z; bfrag[7]  = (_Float16)v4b.w;
        bfrag[8]  = (_Float16)v4c.x; bfrag[9]  = (_Float16)v4c.y;
        bfrag[10] = (_Float16)v4c.z; bfrag[11] = (_Float16)v4c.w;
        bfrag[12] = (_Float16)v4d.x; bfrag[13] = (_Float16)v4d.y;
        bfrag[14] = (_Float16)v4d.z; bfrag[15] = (_Float16)v4d.w;
        v16h a0 = *(const v16h*)(ap0 + (size_t)kc * 32 * 16);
        v16h a1 = *(const v16h*)(ap1 + (size_t)kc * 32 * 16);
        acc0 = __builtin_amdgcn_wmma_f32_16x16x32_f16(false, a0, false, bfrag,
                                                      (short)0, acc0, false, false);
        acc1 = __builtin_amdgcn_wmma_f32_16x16x32_f16(false, a1, false, bfrag,
                                                      (short)0, acc1, false, false);
    }

    if (K % 32 != 0) {                               // single masked tail chunk
        constexpr int kc = FULL;
        const float* sp = src + kc * 32;
        int kb = kc * 32 + lh * 16;
        float xf[16];
#pragma unroll
        for (int j = 0; j < 4; ++j) {
            int eb = kb + 4 * j;
            float4 v4 = make_float4(0.f, 0.f, 0.f, 0.f);
            if (eb + 4 <= K) v4 = *(const float4*)(sp + 4 * j);
            xf[4 * j + 0] = v4.x; xf[4 * j + 1] = v4.y;
            xf[4 * j + 2] = v4.z; xf[4 * j + 3] = v4.w;
        }
        v16h bfrag;
#pragma unroll
        for (int q = 0; q < 16; ++q) bfrag[q] = (_Float16)xf[q];
        v16h a0 = *(const v16h*)(ap0 + (size_t)kc * 32 * 16);
        v16h a1 = *(const v16h*)(ap1 + (size_t)kc * 32 * 16);
        acc0 = __builtin_amdgcn_wmma_f32_16x16x32_f16(false, a0, false, bfrag,
                                                      (short)0, acc0, false, false);
        acc1 = __builtin_amdgcn_wmma_f32_16x16x32_f16(false, a1, false, bfrag,
                                                      (short)0, acc1, false, false);
    }

    if (t < T_DATA) {
        float* dst = out + (size_t)(b * T_DATA + t) * SUB_NO;
#pragma unroll
        for (int r = 0; r < 8; ++r) {
            int s0 = r + 8 * lh;
            dst[s0] = acc0[r];                  // s 0..15
            int s1 = 16 + s0;
            if (s1 < SUB_NO) dst[s1] = acc1[r]; // s 16..19
        }
    }
}

// ---------------------------------------------------------------------------
// Causal FIR, 200 taps: syn_in[b,t,s] = sum_m w0[m]*synE[b,t-m,s]
//                                      + w1[m]*synI[b,t-m,s]
// LDS tiles padded to stride 21 (gcd(21,64)=1 -> conflict-free).
// ---------------------------------------------------------------------------
#define FIR_TT 128
#define FIR_LT (FIR_TT + T_NO - 1)   // 327
__global__ void k_fir(const float* __restrict__ synE,
                      const float* __restrict__ synI,
                      const float* __restrict__ W_syn,
                      float* __restrict__ syn_in)
{
    __shared__ float tE[FIR_LT * 21];
    __shared__ float tI[FIR_LT * 21];
    __shared__ float w0[T_NO], w1[T_NO];

    int b  = blockIdx.y;
    int t0 = blockIdx.x * FIR_TT;
    int tid = threadIdx.x;

    for (int m = tid; m < T_NO; m += blockDim.x) {
        w0[m] = W_syn[2 * m];
        w1[m] = W_syn[2 * m + 1];
    }
    int total = FIR_LT * SUB_NO;
    for (int i = tid; i < total; i += blockDim.x) {
        int tl = i / SUB_NO, s = i % SUB_NO;
        int t = t0 - (T_NO - 1) + tl;
        float ve = 0.f, vi = 0.f;
        if (t >= 0 && t < T_DATA) {
            size_t idx = (size_t)(b * T_DATA + t) * SUB_NO + s;
            ve = synE[idx]; vi = synI[idx];
        }
        tE[tl * 21 + s] = ve;
        tI[tl * 21 + s] = vi;
    }
    __syncthreads();

    int t = t0 + tid;
    if (t < T_DATA) {
        float acc[SUB_NO];
#pragma unroll
        for (int s = 0; s < SUB_NO; ++s) acc[s] = 0.f;
        for (int m = 0; m < T_NO; ++m) {
            float we = w0[m], wi = w1[m];
            int base = (tid + (T_NO - 1) - m) * 21;
#pragma unroll
            for (int s = 0; s < SUB_NO; ++s)
                acc[s] += we * tE[base + s] + wi * tI[base + s];
        }
        float* dst = syn_in + (size_t)(b * T_DATA + t) * SUB_NO;
#pragma unroll
        for (int s = 0; s < SUB_NO; ++s) dst[s] = acc[s];
    }
}

// ---------------------------------------------------------------------------
// Serial recurrence: 8 independent batches -> 8 wave32 waves, lane = subunit.
// leaf_j = sum_k M[j][k] * s_k  (gain folded into M), broadcast via readlane.
// 8-deep register ring prefetch for x_t.
// ---------------------------------------------------------------------------
__global__ void k_scan(const float* __restrict__ syn_in,
                       const float* __restrict__ Mmat,
                       const float* __restrict__ Theta,
                       const float* __restrict__ W_sub,
                       const float* __restrict__ V_o,
                       float* __restrict__ out_final)
{
    int wave = threadIdx.x >> 5;
    int lane = threadIdx.x & 31;
    int b = wave;

    float mr[SUB_NO];
#pragma unroll
    for (int k = 0; k < SUB_NO; ++k)
        mr[k] = (lane < SUB_NO) ? Mmat[lane * SUB_NO + k] : 0.0f;
    float th = (lane < SUB_NO) ? Theta[lane] : 0.0f;
    float g0 = expf(W_sub[0]);
    float vo = V_o[0];

    const float* xin = syn_in + (size_t)b * T_DATA * SUB_NO;
    float s = 0.0f;
    float xb[8];
#pragma unroll
    for (int p = 0; p < 8; ++p)
        xb[p] = (lane < SUB_NO) ? xin[(size_t)p * SUB_NO + lane] : 0.0f;

    for (int tt = 0; tt < T_DATA; tt += 8) {
#pragma unroll
        for (int u = 0; u < 8; ++u) {
            int t = tt + u;
            float x = xb[u];
            int tp = t + 8;
            if (tp < T_DATA)
                xb[u] = (lane < SUB_NO) ? xin[(size_t)tp * SUB_NO + lane] : 0.0f;
            float leaf = 0.0f;
#pragma unroll
            for (int k = 0; k < SUB_NO; ++k)
                leaf += mr[k] * __shfl(s, k, 32);
            s = tanhf(x + th + leaf);
            if (lane == 0)
                out_final[(size_t)b * T_DATA + t] = s * g0 + vo;
        }
    }
}

// ---------------------------------------------------------------------------
extern "C" void kernel_launch(void* const* d_in, const int* in_sizes, int n_in,
                              void* d_out, int out_size, void* d_ws, size_t ws_size,
                              hipStream_t stream)
{
    (void)in_sizes; (void)n_in; (void)out_size; (void)ws_size;

    const float* S_e      = (const float*)d_in[0];
    const float* S_i      = (const float*)d_in[1];
    const float* W_syn    = (const float*)d_in[2];
    const float* W_sub    = (const float*)d_in[3];
    const float* V_o      = (const float*)d_in[4];
    const float* Theta    = (const float*)d_in[5];
    const float* Ce_logit = (const float*)d_in[6];
    const float* Ci_logit = (const float*)d_in[7];
    const float* Cden_raw = (const float*)d_in[8];
    const int*   temp_i   = (const int*)d_in[9];

    float* out = (float*)d_out;
    float* out_final = out;             // [8][5000]   40000
    float* out_filt  = out + 40000;     // [40][200]    8000
    float* out_cden  = out + 48000;     // [20][20]      400
    float* out_ce    = out + 48400;     // [20][2000]  40000
    float* out_ci    = out + 88400;     // [20][500]   10000

    char* ws = (char*)d_ws;
    _Float16* afragE = (_Float16*)(ws);                       // 129024 B
    _Float16* afragI = (_Float16*)(ws + 129024);              //  32768 B
    float* syn_e  = (float*)(ws + 161792);                    // 3.2 MB
    float* syn_i  = (float*)(ws + 161792 + 3200000);          // 3.2 MB
    float* syn_in = (float*)(ws + 161792 + 6400000);          // 3.2 MB
    float* Mmat   = (float*)(ws + 161792 + 9600000);          // 1600 B

    k_softmax_afrag<<<(KCH_E * 32 + 127) / 128, 128, 0, stream>>>(
        Ce_logit, temp_i, out_ce, afragE, E_NO, KCH_E);
    k_softmax_afrag<<<(KCH_I * 32 + 127) / 128, 128, 0, stream>>>(
        Ci_logit, temp_i, out_ci, afragI, I_NO, KCH_I);
    k_setup_den<<<1, 256, 0, stream>>>(
        Cden_raw, W_syn, W_sub, temp_i, out_cden, out_filt, Mmat);

    dim3 gG((T_DATA + 15) / 16, BATCH);
    k_gemm_syn<E_NO, KCH_E><<<gG, 32, 0, stream>>>(S_e, afragE, syn_e);
    k_gemm_syn<I_NO, KCH_I><<<gG, 32, 0, stream>>>(S_i, afragI, syn_i);

    dim3 gF((T_DATA + FIR_TT - 1) / FIR_TT, BATCH);
    k_fir<<<gF, FIR_TT, 0, stream>>>(syn_e, syn_i, W_syn, syn_in);

    k_scan<<<1, 256, 0, stream>>>(syn_in, Mmat, Theta, W_sub, V_o, out_final);
}
